// Chebynet_ATTENTION_Multi_Head_54700703481973
// MI455X (gfx1250) — matrix operand order repaired and checked by reading, not compiled, not run on hardware
//
#include <hip/hip_runtime.h>
#include <hip/hip_bf16.h>

// ---------------------------------------------------------------------------
// Chebynet multi-head attention, MI455X (gfx1250, wave32, WMMA bf16).
//
// B=2048, N=62 (pad 64), C_IN=512, H=8, C_OUT=64, LAYERS=3.
// Block = one batch element; wave w (of 8) = head h.
// Per layer (reassociated):  o_h = relu( a_h @ ((A_hat@x) @ W_h) )
// All GEMMs: v_wmma_f32_16x16x32_bf16 (f32 accumulate).
// All fragment loads are 2x16B vector loads (ds_load_b128 / global_load_b128).
// A_hat tile is DMA'd into LDS by the Tensor Data Mover (with LDS padding to
// a 72-element pitch), guarded so both toolchains compile.
// Workspace: ~3.1 MB.
// ---------------------------------------------------------------------------

typedef __bf16 bf16x16 __attribute__((ext_vector_type(16)));
typedef __bf16 bf16x8  __attribute__((ext_vector_type(8)));
typedef __bf16 bf16x4  __attribute__((ext_vector_type(4)));
typedef float  v8f     __attribute__((ext_vector_type(8)));
typedef unsigned int u32x4 __attribute__((ext_vector_type(4)));
typedef int          i32x4 __attribute__((ext_vector_type(4)));
typedef int          i32x8 __attribute__((ext_vector_type(8)));

#define NPAD   64
#define CIN    512
#define CHK    128
#define XPITCH 136            // bf16/row: 272 B = 17x16B -> aligned + bank-spread
#define TPITCH 72             // 64-wide tiles: 144 B = 9x16B
#define NHEAD  8
#define NLAY   3
#define NSEQ   62
#define BATCH  2048

// ---- LDS layout (staging unions chunk buffers; guarded by barriers) --------
#define SM_XS    0                          // bf16 [64][XPITCH]
#define SM_AXS   17408                      // bf16 [64][XPITCH]
#define SM_XST   34816                      // bf16 [128][TPITCH]   (x chunk, T)
#define SM_STAGE 0                          // per wave 18432 B: a | axw^T
#define SM_AHAT  147456                     // bf16 [64][TPITCH]
#define SM_DNEG  156672                     // f32 [64]
#define SM_BYTES 156928                     // < 160 KB

#if defined(__gfx1250__) && __has_builtin(__builtin_amdgcn_tensor_load_to_lds) \
    && __has_builtin(__builtin_amdgcn_s_wait_tensorcnt)
#define USE_TDM 1
#else
#define USE_TDM 0
#endif

// ---------------- fragment loaders (ISA 7.12.2 layouts, wave32) -------------
__device__ __forceinline__ bf16x16 cat16(bf16x8 lo, bf16x8 hi) {
  return __builtin_shufflevector(lo, hi, 0, 1, 2, 3, 4, 5, 6, 7,
                                 8, 9, 10, 11, 12, 13, 14, 15);
}
// A (16x32): lane = M; lane>>4 picks K-halves {0..7,16..23} / {8..15,24..31}.
// Both 8-element runs are contiguous -> two 16B LDS loads.
__device__ __forceinline__ bf16x16 load_a_frag(const __bf16* p, int ld) {
  const int lane = threadIdx.x & 31;
  const __bf16* q = p + (lane & 15) * ld + ((lane >> 4) << 3);
  return cat16(*(const bf16x8*)q, *(const bf16x8*)(q + 16));
}
// B (32x16) from [N][K] (transposed) storage: lane = N; lane>>4 picks K-halves {0..7,16..23} / {8..15,24..31}.
__device__ __forceinline__ bf16x16 load_bt_frag(const __bf16* p, int ld) {
  const int lane = threadIdx.x & 31;
  const __bf16* q = p + (lane & 15) * ld + ((lane >> 4) << 3);
  return cat16(*(const bf16x8*)q, *(const bf16x8*)(q + 16));
}

__device__ __forceinline__ v8f wmma_bf16(bf16x16 a, bf16x16 b, v8f c) {
  return __builtin_amdgcn_wmma_f32_16x16x32_bf16(false, a, false, b,
                                                 (short)0, c, false, false);
}
// xor-16 lane swap (wave32): ds_swizzle group-of-32, xor=0x10, and=0x1f
__device__ __forceinline__ float xor16(float v) {
  return __int_as_float(__builtin_amdgcn_ds_swizzle(__float_as_int(v), 0x401F));
}

// ---------------------------------------------------------------------------
// Prep 1: adjacency powers -> A_hat (bf16, dense 64x64) + d_neg.
// ---------------------------------------------------------------------------
__global__ void prep_adj(const float* __restrict__ L,
                         __bf16* __restrict__ ahat_g,
                         float* __restrict__ dneg_g) {
  __shared__ float A2s[NSEQ * NSEQ];
  const int tid = threadIdx.x;
  for (int idx = tid; idx < NSEQ * NSEQ; idx += 256) {
    const int n = idx / NSEQ, m = idx % NSEQ;
    float s = 0.f;
    for (int k = 0; k < NSEQ; ++k) s += L[n * NSEQ + k] * L[k * NSEQ + m];
    A2s[idx] = s;
  }
  __syncthreads();
  for (int idx = tid; idx < NLAY * NPAD * NPAD; idx += 256) {
    const int l = idx >> 12, n = (idx >> 6) & 63, m = idx & 63;
    float a = 0.f;
    if (n < NSEQ && m < NSEQ)
      a = (l == 0) ? (n == m ? 1.f : 0.f)
                   : (l == 1 ? L[n * NSEQ + m] : A2s[n * NSEQ + m]);
    ahat_g[idx] = (__bf16)(a + ((n == m && n < NSEQ) ? 1.f : 0.f));
  }
  for (int idx = tid; idx < NLAY * NPAD; idx += 256) {
    const int l = idx >> 6, n = idx & 63;
    float dv = 0.f;
    if (n < NSEQ) {
      float D = 0.f;
      for (int m = 0; m < NSEQ; ++m)
        D += (l == 0) ? (n == m ? 1.f : 0.f)
                      : (l == 1 ? L[n * NSEQ + m] : A2s[n * NSEQ + m]);
      dv = (D == 0.f) ? 0.f : 1.f / D;
    }
    dneg_g[idx] = dv;
  }
}

// ---------------------------------------------------------------------------
// Prep 2: weights -> bf16, B-fragment layout [l][h][n][k].
// ---------------------------------------------------------------------------
__global__ void prep_w(const float* __restrict__ Wa, const float* __restrict__ W,
                       __bf16* __restrict__ Wat, __bf16* __restrict__ Wt2) {
  const long idx = (long)blockIdx.x * 256 + threadIdx.x;   // 3*8*64*512
  if (idx >= (long)NLAY * NHEAD * NPAD * CIN) return;
  const int c  = idx & (CIN - 1);
  const int m  = (idx >> 9) & 63;
  const int lh = (int)(idx >> 15);
  Wat[idx] = (__bf16)((m < NSEQ) ? Wa[((long)lh * CIN + c) * NSEQ + m] : 0.f);
  Wt2[idx] = (__bf16)W[((long)lh * CIN + c) * NPAD + m];
}

// ---------------------------------------------------------------------------
// Main fused kernel: 2048 blocks x 256 threads (8 wave32 = 8 heads).
// ---------------------------------------------------------------------------
__global__ __launch_bounds__(256, 1)
void cheb_attn(const float* __restrict__ x,
               const __bf16* __restrict__ ahat_g,
               const float* __restrict__ dneg_g,
               const __bf16* __restrict__ Wat,
               const __bf16* __restrict__ Wt2,
               float* __restrict__ out) {
  __shared__ __attribute__((aligned(16))) unsigned char smem[SM_BYTES];
  __bf16* xs   = (__bf16*)(smem + SM_XS);    // [64][XPITCH]  row-major x chunk
  __bf16* axs  = (__bf16*)(smem + SM_AXS);   // [64][XPITCH]  A_hat@x chunk
  __bf16* xsT  = (__bf16*)(smem + SM_XST);   // [128][TPITCH] x chunk transposed
  __bf16* ahat = (__bf16*)(smem + SM_AHAT);  // [64][TPITCH]
  float*  dneg = (float*)(smem + SM_DNEG);

  const int tid  = threadIdx.x;
  const int lane = tid & 31;
  const int w    = tid >> 5;                 // wave id == head id
  const int nlo  = lane & 15;                // tile column (C layout)
  const int rhi  = (lane >> 4) << 3;         // +8 rows for lanes 16..31
  __bf16* stg_a    = (__bf16*)(smem + SM_STAGE + w * 18432);          // [64][72]
  __bf16* stg_axwT = (__bf16*)(smem + SM_STAGE + w * 18432 + 9216);   // [64][72]
  const int b = blockIdx.x;

  v8f OUT[4][4];
#pragma unroll
  for (int i = 0; i < 4; ++i)
#pragma unroll
    for (int j = 0; j < 4; ++j) OUT[i][j] = {};

#pragma unroll 1
  for (int l = 0; l < NLAY; ++l) {
    __syncthreads();                         // prev-layer staging readers done
    // ---- A_hat tile -> LDS via Tensor Data Mover (pad 4dw per 32dw => pitch
    //      72 elements); fallback: plain cooperative copy. Static LDS block is
    //      at offset 0 of the wave's LDS allocation.
#if USE_TDM
    if (tid < 32) {                          // wave 0 issues one TDM descriptor
      const unsigned long long ga =
          (unsigned long long)(const void*)(ahat_g + l * NPAD * NPAD);
      u32x4 g0;
      g0[0] = 1u;                                        // count=1 (user D#)
      g0[1] = (unsigned)SM_AHAT;                         // lds_addr
      g0[2] = (unsigned)(ga & 0xFFFFFFFFu);              // global_addr[95:64]
      g0[3] = (unsigned)((ga >> 32) & 0x1FFFFFFu) | (2u << 30);  // type=2
      i32x8 g1 = {};
      g1[0] = (1 << 16) | (1 << 20) | (4 << 22) | (3 << 25);
      //        ^2B elems  ^pad_en    ^every 32dw ^pad 4dw
      g1[1] = (int)(64u << 16);        // tensor_dim0 = 64
      g1[2] = (int)(64u << 16);        // tensor_dim1 = 64
      g1[3] = (int)(64u << 16);        // tile_dim0   = 64
      g1[4] = 64;                      // tile_dim1   = 64
      g1[5] = 64;                      // tensor_dim0_stride = 64
      i32x4 gz = {};
#if __has_include(<hip/amd_detail/amd_gfx1250_TDM.h>)
      i32x8 gz8 = {};
      __builtin_amdgcn_tensor_load_to_lds(g0, g1, gz, gz, gz8, 0);
#else
      __builtin_amdgcn_tensor_load_to_lds(g0, g1, gz, gz, 0);
#endif
      __builtin_amdgcn_s_wait_tensorcnt(0);
    }
#else
    for (int i = tid; i < NPAD * NPAD; i += 256)
      ahat[(i >> 6) * TPITCH + (i & 63)] = ahat_g[l * NPAD * NPAD + i];
#endif
    if (tid < NPAD) dneg[tid] = dneg_g[l * NPAD + tid];

    v8f S[4][4], AXW[4][4];
#pragma unroll
    for (int i = 0; i < 4; ++i)
#pragma unroll
      for (int j = 0; j < 4; ++j) { S[i][j] = {}; AXW[i][j] = {}; }

#pragma unroll 1
    for (int kc = 0; kc < CIN / CHK; ++kc) {
      __syncthreads();                       // prev chunk readers done
      // ---- stage x[b][:, kc*CHK..+CHK] as bf16: row-major + transposed ----
      for (int i = tid; i < NPAD * CHK / 4; i += 256) {
        const int row = i >> 5;              // 32 float4 per row
        const int c4  = (i & 31) << 2;
        float4 v = make_float4(0.f, 0.f, 0.f, 0.f);
        if (row < NSEQ) {
          const float* gp = &x[((size_t)b * NSEQ + row) * CIN + kc * CHK + c4];
          v = *(const float4*)gp;
          if (kc + 1 < CIN / CHK) __builtin_prefetch(gp + CHK, 0, 0);
        }
        bf16x4 bv = {(__bf16)v.x, (__bf16)v.y, (__bf16)v.z, (__bf16)v.w};
        *(bf16x4*)&xs[row * XPITCH + c4] = bv;           // 8B packed store
        xsT[(c4 + 0) * TPITCH + row] = bv[0];            // transposed scatter
        xsT[(c4 + 1) * TPITCH + row] = bv[1];
        xsT[(c4 + 2) * TPITCH + row] = bv[2];
        xsT[(c4 + 3) * TPITCH + row] = bv[3];
      }
      __syncthreads();
      // ---- ax chunk = A_hat @ x; wave w owns column tile w ----------------
#pragma unroll
      for (int Mt = 0; Mt < 4; ++Mt) {
        v8f acc = {};
#pragma unroll
        for (int ks = 0; ks < 2; ++ks) {
          bf16x16 af = load_a_frag(ahat + (Mt * 16) * TPITCH + ks * 32, TPITCH);
          bf16x16 bf = load_bt_frag(xsT + (w * 16) * TPITCH + ks * 32, TPITCH);
          acc = wmma_bf16(af, bf, acc);
        }
#pragma unroll
        for (int r = 0; r < 8; ++r)
          axs[(Mt * 16 + rhi + r) * XPITCH + w * 16 + nlo] = (__bf16)acc[r];
      }
      __syncthreads();
      // ---- S += x@Walpha ; AXW += ax@W  (K = this chunk) ------------------
      const __bf16* WatB = Wat + ((size_t)(l * NHEAD + w) * NPAD) * CIN + kc * CHK;
      const __bf16* Wt2B = Wt2 + ((size_t)(l * NHEAD + w) * NPAD) * CIN + kc * CHK;
#pragma unroll
      for (int ks = 0; ks < CHK / 32; ++ks) {
        bf16x16 aX[4], aAX[4];
#pragma unroll
        for (int Mt = 0; Mt < 4; ++Mt) {
          aX[Mt]  = load_a_frag(xs  + (Mt * 16) * XPITCH + ks * 32, XPITCH);
          aAX[Mt] = load_a_frag(axs + (Mt * 16) * XPITCH + ks * 32, XPITCH);
        }
#pragma unroll
        for (int Nt = 0; Nt < 4; ++Nt) {
          bf16x16 bw  = load_bt_frag(WatB + (size_t)(Nt * 16) * CIN + ks * 32, CIN);
          bf16x16 bw2 = load_bt_frag(Wt2B + (size_t)(Nt * 16) * CIN + ks * 32, CIN);
#pragma unroll
          for (int Mt = 0; Mt < 4; ++Mt) {
            S[Mt][Nt]   = wmma_bf16(aX[Mt],  bw,  S[Mt][Nt]);
            AXW[Mt][Nt] = wmma_bf16(aAX[Mt], bw2, AXW[Mt][Nt]);
          }
        }
      }
    }
    __syncthreads();   // chunk buffers dead -> per-wave staging may reuse LDS

    // ---- attention: d_neg scale, leaky relu, softmax over rows (axis=-2) --
#pragma unroll
    for (int Mt = 0; Mt < 4; ++Mt)
#pragma unroll
      for (int Nt = 0; Nt < 4; ++Nt)
#pragma unroll
        for (int r = 0; r < 8; ++r) {
          const int row = Mt * 16 + rhi + r;
          float v = S[Mt][Nt][r] * dneg[row];
          v = v > 0.f ? v : 0.01f * v;
          if (row >= NSEQ) v = -1.0e30f;     // mask padded rows
          S[Mt][Nt][r] = v;
        }
#pragma unroll
    for (int Nt = 0; Nt < 4; ++Nt) {
      float mx = -3.0e38f;
#pragma unroll
      for (int Mt = 0; Mt < 4; ++Mt)
#pragma unroll
        for (int r = 0; r < 8; ++r) mx = fmaxf(mx, S[Mt][Nt][r]);
      mx = fmaxf(mx, xor16(mx));             // lanes j <-> j+16: full column
      float sm = 0.f;
#pragma unroll
      for (int Mt = 0; Mt < 4; ++Mt)
#pragma unroll
        for (int r = 0; r < 8; ++r) {
          const float e = __expf(S[Mt][Nt][r] - mx);
          S[Mt][Nt][r] = e;
          sm += e;
        }
      sm += xor16(sm);
      const float inv = 1.0f / sm;
#pragma unroll
      for (int Mt = 0; Mt < 4; ++Mt)
#pragma unroll
        for (int r = 0; r < 8; ++r) S[Mt][Nt][r] *= inv;
    }
    // ---- stage a (row-major) and AXW (transposed) as bf16 -----------------
#pragma unroll
    for (int Mt = 0; Mt < 4; ++Mt)
#pragma unroll
      for (int Nt = 0; Nt < 4; ++Nt)
#pragma unroll
        for (int r = 0; r < 8; ++r) {
          const int row = Mt * 16 + rhi + r, col = Nt * 16 + nlo;
          stg_a[row * TPITCH + col]    = (__bf16)S[Mt][Nt][r];
          stg_axwT[col * TPITCH + row] = (__bf16)AXW[Mt][Nt][r];
        }
    // ---- o = relu(a @ AXW), accumulate over layers ------------------------
#pragma unroll
    for (int Mt = 0; Mt < 4; ++Mt)
#pragma unroll
      for (int Nt = 0; Nt < 4; ++Nt) {
        v8f o2 = {};
#pragma unroll
        for (int ks = 0; ks < 2; ++ks) {
          bf16x16 aa = load_a_frag(stg_a + (Mt * 16) * TPITCH + ks * 32, TPITCH);
          bf16x16 bb = load_bt_frag(stg_axwT + (Nt * 16) * TPITCH + ks * 32, TPITCH);
          o2 = wmma_bf16(aa, bb, o2);
        }
#pragma unroll
        for (int r = 0; r < 8; ++r) OUT[Mt][Nt][r] += fmaxf(o2[r], 0.f);
      }
  }

  // final relu is identity (sum of relu terms >= 0); store rows < 62
#pragma unroll
  for (int Mt = 0; Mt < 4; ++Mt)
#pragma unroll
    for (int Nt = 0; Nt < 4; ++Nt)
#pragma unroll
      for (int r = 0; r < 8; ++r) {
        const int row = Mt * 16 + rhi + r;
        if (row < NSEQ)
          out[((size_t)blockIdx.x * NSEQ + row) * CIN + w * 64 + Nt * 16 + nlo] =
              OUT[Mt][Nt][r];
      }
}

// ---------------------------------------------------------------------------
extern "C" void kernel_launch(void* const* d_in, const int* in_sizes, int n_in,
                              void* d_out, int out_size, void* d_ws, size_t ws_size,
                              hipStream_t stream) {
  const float* x  = (const float*)d_in[0];   // (2048, 62, 512)
  const float* L  = (const float*)d_in[1];   // (62, 62)
  const float* Wa = (const float*)d_in[2];   // (3, 8, 512, 62)
  const float* W  = (const float*)d_in[3];   // (3, 8, 512, 64)

  unsigned char* ws = (unsigned char*)d_ws;  // needs ~3.1 MB
  __bf16* ahat_g = (__bf16*)(ws);                       // 3*64*64*2 = 24576
  float*  dneg_g = (float*)(ws + 24576);                // 3*64*4    = 768
  __bf16* Wat    = (__bf16*)(ws + 25600);               // 3*8*64*512*2
  __bf16* Wt2    = (__bf16*)(ws + 25600 + 1572864);     // 3*8*64*512*2

  prep_adj<<<1, 256, 0, stream>>>(L, ahat_g, dneg_g);
  prep_w<<<(NLAY * NHEAD * NPAD * CIN) / 256, 256, 0, stream>>>(Wa, W, Wat, Wt2);
  cheb_attn<<<BATCH, 256, 0, stream>>>(x, ahat_g, dneg_g, Wat, Wt2,
                                       (float*)d_out);
}
